// NodeModel_24773371363898
// MI455X (gfx1250) — matrix operand order repaired and checked by reading, not compiled, not run on hardware
//
#include <hip/hip_runtime.h>
#include <hip/hip_bf16.h>

// ---------------------------------------------------------------------------
// GraphNet NodeModel on MI455X (gfx1250, wave32, WMMA).
// Edge MLP -> scatter-mean (f32 atomics, L2-resident accumulator) -> Node MLP.
// f16 WMMA (f32 accumulate): 26 GFLOP of GEMM sits far below the ~23us HBM
// floor (0.53 GB @ 23.3 TB/s), so the kernel should be memory-bound.
// All WMMA fragments are fetched as ds_load_b128 pairs:
//   - A fragments: contiguous k order, two adjacent 8-half runs
//     (K = 16h..16h+7 and K = 16h+8..16h+15 within each 32-wide K block).
//   - B fragments: weights are pre-swizzled once per block into a
//     fragment-major LDS layout (32 contiguous bytes per lane per fragment).
// ---------------------------------------------------------------------------

typedef __attribute__((ext_vector_type(16))) _Float16 v16h;
typedef __attribute__((ext_vector_type(8)))  _Float16 v8h;
typedef __attribute__((ext_vector_type(4)))  _Float16 v4h;
typedef __attribute__((ext_vector_type(2)))  _Float16 v2h;
typedef __attribute__((ext_vector_type(8)))  float    v8f;

#define N_NODES 50000
#define N_EDGES 1000000

__device__ __forceinline__ v16h cat16(v8h lo, v8h hi) {
    return __builtin_shufflevector(lo, hi, 0, 1, 2, 3, 4, 5, 6, 7,
                                   8, 9, 10, 11, 12, 13, 14, 15);
}

// A-fragment for K-block kb from a 16B-aligned LDS row (row = A[m][*]).
__device__ __forceinline__ v16h load_afrag(const _Float16* row, int kb, int h) {
    v8h lo = *(const v8h*)(row + kb * 32 + 16 * h);
    v8h hi = *(const v8h*)(row + kb * 32 + 16 * h + 8);
    return cat16(lo, hi);
}

// B-fragment from fragment-major swizzled weights: 32 contiguous bytes/lane.
__device__ __forceinline__ v16h load_bfrag(const _Float16* base, int f, int lane) {
    const _Float16* p = base + ((size_t)f * 32 + lane) * 16;
    v8h lo = *(const v8h*)p;
    v8h hi = *(const v8h*)(p + 8);
    return cat16(lo, hi);
}

__device__ __forceinline__ void lds_fence() {
    asm volatile("s_wait_dscnt 0x0" ::: "memory");
}

// ------------------------- edge pass + scatter -----------------------------
__global__ __launch_bounds__(256) void edge_kernel(
    const float* __restrict__ x, const int* __restrict__ eidx,
    const float* __restrict__ ea,
    const float* __restrict__ W1, const float* __restrict__ b1,
    const float* __restrict__ W2, const float* __restrict__ b2,
    float* __restrict__ sums, float* __restrict__ counts)
{
    // fragment-major W1: frag f = kb*4+nt, per lane 16 contiguous halves
    __shared__ __attribute__((aligned(16))) _Float16 sW1f[128 * 64];   // 16 KB
    __shared__ __attribute__((aligned(16))) _Float16 sA[8][16][136];   // 34.8 KB
    __shared__ __attribute__((aligned(16))) _Float16 sH[8][16][72];    // 18.4 KB
    __shared__ int sCol[8][16];

    const int tid  = threadIdx.x;
    const int wave = tid >> 5;
    const int lane = tid & 31;
    const int ln   = lane & 15;   // row m for A, column n for B/C
    const int h    = lane >> 4;

    // swizzle W1 -> fragment-major LDS (once per block)
    for (int i = tid; i < 128 * 64; i += 256) {
        int j = i & 15, li = (i >> 4) & 31, f = i >> 9;
        int K = (f >> 2) * 32 + j + 16 * (li >> 4);
        int N = (f & 3) * 16 + (li & 15);
        sW1f[i] = (_Float16)W1[K * 64 + N];
    }
    __syncthreads();

    // W2 fragments live in registers for the whole kernel (8 frags = 64 VGPR)
    v16h w2f[2][4];
    #pragma unroll
    for (int kb = 0; kb < 2; ++kb)
        #pragma unroll
        for (int nt = 0; nt < 4; ++nt)
            #pragma unroll
            for (int j = 0; j < 16; ++j)
                w2f[kb][nt][j] =
                    (_Float16)W2[(kb * 32 + j + 16 * h) * 64 + nt * 16 + ln];

    const int nTiles = N_EDGES / 16;            // 62500
    for (int tile = blockIdx.x * 8 + wave; tile < nTiles;
         tile += gridDim.x * 8) {
        const int ebase = tile * 16;

        int rowi = 0, coli = 0;
        if (lane < 16) {
            rowi = eidx[ebase + lane];
            coli = eidx[N_EDGES + ebase + lane];
            sCol[wave][lane] = coli;
        }

        // stage A = [x[row] || edge_attr] as 16x128 f16 (one b64 store/lane/row)
        #pragma unroll 4
        for (int m = 0; m < 16; ++m) {
            int rm = __shfl(rowi, m, 32);
            float4 v;
            if (lane < 16)
                v = *(const float4*)&x[(size_t)rm * 64 + lane * 4];
            else
                v = *(const float4*)&ea[(size_t)(ebase + m) * 64 + (lane - 16) * 4];
            int k0 = (lane < 16) ? lane * 4 : 64 + (lane - 16) * 4;
            v4h hv = {(_Float16)v.x, (_Float16)v.y, (_Float16)v.z, (_Float16)v.w};
            *(v4h*)&sA[wave][m][k0] = hv;
        }
        lds_fence();

        // layer 1: 16x128 @ 128x64
        v8f c1[4];
        #pragma unroll
        for (int nt = 0; nt < 4; ++nt)
            #pragma unroll
            for (int r = 0; r < 8; ++r) c1[nt][r] = 0.0f;

        const _Float16* arow = &sA[wave][ln][0];
        #pragma unroll
        for (int kb = 0; kb < 4; ++kb) {
            v16h a = load_afrag(arow, kb, h);
            #pragma unroll
            for (int nt = 0; nt < 4; ++nt) {
                v16h b = load_bfrag(sW1f, kb * 4 + nt, lane);
                c1[nt] = __builtin_amdgcn_wmma_f32_16x16x32_f16(
                    false, a, false, b, (short)0, c1[nt], false, false);
            }
        }

        // bias + ReLU -> hidden (16x64 f16) in LDS (layout change C -> A)
        #pragma unroll
        for (int nt = 0; nt < 4; ++nt) {
            float bias = b1[nt * 16 + ln];
            #pragma unroll
            for (int r = 0; r < 8; ++r) {
                float vv = fmaxf(c1[nt][r] + bias, 0.0f);
                sH[wave][r + 8 * h][nt * 16 + ln] = (_Float16)vv;
            }
        }
        lds_fence();

        // layer 2: 16x64 @ 64x64 (B operand in registers)
        v8f c2[4];
        #pragma unroll
        for (int nt = 0; nt < 4; ++nt)
            #pragma unroll
            for (int r = 0; r < 8; ++r) c2[nt][r] = 0.0f;

        const _Float16* hrow = &sH[wave][ln][0];
        #pragma unroll
        for (int kb = 0; kb < 2; ++kb) {
            v16h a = load_afrag(hrow, kb, h);
            #pragma unroll
            for (int nt = 0; nt < 4; ++nt)
                c2[nt] = __builtin_amdgcn_wmma_f32_16x16x32_f16(
                    false, a, false, w2f[kb][nt], (short)0, c2[nt], false, false);
        }

        // bias + scatter-add into sums (f32 atomics, L2-resident)
        #pragma unroll
        for (int nt = 0; nt < 4; ++nt) {
            float bias = b2[nt * 16 + ln];
            #pragma unroll
            for (int r = 0; r < 8; ++r) {
                int M = r + 8 * h;
                int cnode = sCol[wave][M];
                atomicAdd(&sums[(size_t)cnode * 64 + nt * 16 + ln],
                          c2[nt][r] + bias);
            }
        }
        if (lane < 16) atomicAdd(&counts[coli], 1.0f);
    }
}

// ----------------------------- node pass -----------------------------------
__global__ __launch_bounds__(256) void node_kernel(
    const float* __restrict__ x, const float* __restrict__ u,
    const int* __restrict__ batch,
    const float* __restrict__ W3, const float* __restrict__ b3,
    const float* __restrict__ W4, const float* __restrict__ b4,
    const float* __restrict__ sums, const float* __restrict__ counts,
    float* __restrict__ out)
{
    __shared__ __attribute__((aligned(16))) _Float16 sW3f[192 * 64];   // 24 KB
    __shared__ __attribute__((aligned(16))) _Float16 sA[8][16][200];   // 51.2 KB
    __shared__ __attribute__((aligned(16))) _Float16 sH[8][16][72];    // 18.4 KB

    const int tid  = threadIdx.x;
    const int wave = tid >> 5;
    const int lane = tid & 31;
    const int ln   = lane & 15;
    const int h    = lane >> 4;

    for (int i = tid; i < 192 * 64; i += 256) {
        int j = i & 15, li = (i >> 4) & 31, f = i >> 9;
        int K = (f >> 2) * 32 + j + 16 * (li >> 4);
        int N = (f & 3) * 16 + (li & 15);
        sW3f[i] = (_Float16)W3[K * 64 + N];
    }
    __syncthreads();

    v16h w4f[2][4];
    #pragma unroll
    for (int kb = 0; kb < 2; ++kb)
        #pragma unroll
        for (int nt = 0; nt < 4; ++nt)
            #pragma unroll
            for (int j = 0; j < 16; ++j)
                w4f[kb][nt][j] =
                    (_Float16)W4[(kb * 32 + j + 16 * h) * 64 + nt * 16 + ln];

    const int nTiles = N_NODES / 16;            // 3125
    for (int tile = blockIdx.x * 8 + wave; tile < nTiles;
         tile += gridDim.x * 8) {
        const int nbase = tile * 16;

        float invc = 0.0f; int bm = 0;
        if (lane < 16) {
            invc = 1.0f / fmaxf(counts[nbase + lane], 1.0f);
            bm   = batch[nbase + lane];
        }

        // stage A = [x || sums*inv || u[batch]] as 16x192 f16 (b32 stores)
        #pragma unroll 2
        for (int m = 0; m < 16; ++m) {
            float sc = __shfl(invc, m, 32);
            int   ub = __shfl(bm,   m, 32);
            float2 vx = *(const float2*)&x[(size_t)(nbase + m) * 64 + lane * 2];
            float2 vs = *(const float2*)&sums[(size_t)(nbase + m) * 64 + lane * 2];
            float2 vu = *(const float2*)&u[(size_t)ub * 64 + lane * 2];
            _Float16* rowp = &sA[wave][m][0];
            ((v2h*)(rowp))[lane]       = (v2h){(_Float16)vx.x, (_Float16)vx.y};
            ((v2h*)(rowp + 64))[lane]  = (v2h){(_Float16)(vs.x * sc),
                                               (_Float16)(vs.y * sc)};
            ((v2h*)(rowp + 128))[lane] = (v2h){(_Float16)vu.x, (_Float16)vu.y};
        }
        lds_fence();

        // layer 1: 16x192 @ 192x64
        v8f c1[4];
        #pragma unroll
        for (int nt = 0; nt < 4; ++nt)
            #pragma unroll
            for (int r = 0; r < 8; ++r) c1[nt][r] = 0.0f;

        const _Float16* arow = &sA[wave][ln][0];
        #pragma unroll
        for (int kb = 0; kb < 6; ++kb) {
            v16h a = load_afrag(arow, kb, h);
            #pragma unroll
            for (int nt = 0; nt < 4; ++nt) {
                v16h b = load_bfrag(sW3f, kb * 4 + nt, lane);
                c1[nt] = __builtin_amdgcn_wmma_f32_16x16x32_f16(
                    false, a, false, b, (short)0, c1[nt], false, false);
            }
        }

        #pragma unroll
        for (int nt = 0; nt < 4; ++nt) {
            float bias = b3[nt * 16 + ln];
            #pragma unroll
            for (int r = 0; r < 8; ++r) {
                float vv = fmaxf(c1[nt][r] + bias, 0.0f);
                sH[wave][r + 8 * h][nt * 16 + ln] = (_Float16)vv;
            }
        }
        lds_fence();

        // layer 2: 16x64 @ 64x64 (B in registers)
        v8f c2[4];
        #pragma unroll
        for (int nt = 0; nt < 4; ++nt)
            #pragma unroll
            for (int r = 0; r < 8; ++r) c2[nt][r] = 0.0f;

        const _Float16* hrow = &sH[wave][ln][0];
        #pragma unroll
        for (int kb = 0; kb < 2; ++kb) {
            v16h a = load_afrag(hrow, kb, h);
            #pragma unroll
            for (int nt = 0; nt < 4; ++nt)
                c2[nt] = __builtin_amdgcn_wmma_f32_16x16x32_f16(
                    false, a, false, w4f[kb][nt], (short)0, c2[nt], false, false);
        }

        #pragma unroll
        for (int nt = 0; nt < 4; ++nt) {
            float bias = b4[nt * 16 + ln];
            #pragma unroll
            for (int r = 0; r < 8; ++r) {
                int M = r + 8 * h;
                out[(size_t)(nbase + M) * 64 + nt * 16 + ln] = c2[nt][r] + bias;
            }
        }
    }
}

// ---------------------------------------------------------------------------
extern "C" void kernel_launch(void* const* d_in, const int* in_sizes, int n_in,
                              void* d_out, int out_size, void* d_ws, size_t ws_size,
                              hipStream_t stream) {
    const float* x     = (const float*)d_in[0];
    const int*   eidx  = (const int*)  d_in[1];
    const float* ea    = (const float*)d_in[2];
    const float* u     = (const float*)d_in[3];
    const int*   batch = (const int*)  d_in[4];
    const float* W1 = (const float*)d_in[5];  const float* b1 = (const float*)d_in[6];
    const float* W2 = (const float*)d_in[7];  const float* b2 = (const float*)d_in[8];
    const float* W3 = (const float*)d_in[9];  const float* b3 = (const float*)d_in[10];
    const float* W4 = (const float*)d_in[11]; const float* b4 = (const float*)d_in[12];
    float* out = (float*)d_out;

    float* sums   = (float*)d_ws;                       // [N_NODES * 64]
    float* counts = sums + (size_t)N_NODES * 64;        // [N_NODES]

    hipMemsetAsync(d_ws, 0, ((size_t)N_NODES * 64 + N_NODES) * sizeof(float),
                   stream);

    edge_kernel<<<1024, 256, 0, stream>>>(x, eidx, ea, W1, b1, W2, b2,
                                          sums, counts);
    node_kernel<<<256, 256, 0, stream>>>(x, u, batch, W3, b3, W4, b4,
                                         sums, counts, out);
}
